// seq2seqTrain_66735201845868
// MI455X (gfx1250) — matrix operand order repaired and checked by reading, not compiled, not run on hardware
//
#include <hip/hip_runtime.h>
#include <hip/hip_bf16.h>
#include <math.h>

// ---------------------------------------------------------------------------
// Seq2seq GRU + attention forward loss for MI455X (gfx1250, wave32).
// All large matmuls use V_WMMA_F32_16X16X32_BF16 (bf16 inputs, f32 accum).
// Fragments are loaded as aligned b128 chunks from LDS; B tiles are staged
// transposed so B-fragment loads are contiguous.
// ---------------------------------------------------------------------------

typedef __attribute__((ext_vector_type(16))) __bf16 v16bf;
typedef __attribute__((ext_vector_type(8)))  float  v8f;
typedef __attribute__((ext_vector_type(4)))  unsigned int u32x4;
typedef unsigned short u16t;

#define B_ 64
#define S_ 128
#define T_ 128
#define E_ 256
#define U_ 1024
#define V_ 8000

union FragBF { v16bf v; u16t s[16]; u32x4 q[2]; };
union FragF  { v8f  v; float f[8];  };
union Pack8  { u32x4 q; u16t s[8]; };

__device__ inline u16t f2bf(float f) {
  unsigned u = __float_as_uint(f);
  unsigned r = 0x7FFFu + ((u >> 16) & 1u);   // round-to-nearest-even
  return (u16t)((u + r) >> 16);
}

__device__ inline v8f wmma_bf16(v16bf a, v16bf b, v8f c) {
  // 8 args: (neg_a, A, neg_b, B, c_mod, C, reuse_a, reuse_b)
  return __builtin_amdgcn_wmma_f32_16x16x32_bf16(false, a, false, b,
                                                 (short)0, c, false, false);
}

// A fragment from row-major LDS tile (stride 32 elems = 64B rows).
// ISA layout: lane<16 -> K {half*8+0..7, 16+half*8+0..7}; two aligned b128.
__device__ inline v16bf frag_a_lds(const u16t* As, int row0) {
  FragBF f;
  int lane = threadIdx.x & 31, half = lane >> 4, r = lane & 15;
  const u16t* base = As + (row0 + r) * 32 + half * 8;
  f.q[0] = *(const u32x4*)(base);        // K = half*8 + 0..7
  f.q[1] = *(const u32x4*)(base + 16);   // K = 16 + half*8 + 0..7
  return f.v;
}

// B fragment from transposed LDS tile Bt[col][k] (stride 32 elems).
// Lane col = lane&15, K {half*8+0..7, 16+half*8+0..7} -> two b128.
__device__ inline v16bf frag_b_lds(const u16t* Bt, int col0) {
  FragBF f;
  int lane = threadIdx.x & 31, half = lane >> 4, c = lane & 15;
  const u16t* base = Bt + (col0 + c) * 32 + half * 8;
  f.q[0] = *(const u32x4*)(base);
  f.q[1] = *(const u32x4*)(base + 16);
  return f.v;
}

// ------------------------- elementwise helpers -----------------------------

__global__ void k_cvt(const float* s, u16t* d, int n) {
  int i = blockIdx.x * 256 + threadIdx.x;
  if (i < n) d[i] = f2bf(s[i]);
}
__global__ void k_zero_f32(float* p, int n) {
  int i = blockIdx.x * 256 + threadIdx.x;
  if (i < n) p[i] = 0.f;
}
__global__ void k_zero_u16(u16t* p, int n) {
  int i = blockIdx.x * 256 + threadIdx.x;
  if (i < n) p[i] = 0;
}
__global__ void k_gather(const int* tok, const float* emb, u16t* X,
                         int rows, int L, int tokStride) {
  int i = blockIdx.x * 256 + threadIdx.x;
  if (i >= rows * E_) return;
  int row = i / E_, e = i % E_;
  int b = row / L, t = row % L;
  int tk = tok[b * tokStride + t];
  X[i] = f2bf(emb[(size_t)tk * E_ + e]);
}
__global__ void k_tanh_bf(const float* x, u16t* y, int n) {
  int i = blockIdx.x * 256 + threadIdx.x;
  if (i < n) y[i] = f2bf(tanhf(x[i]));
}

// ------------------------- generic bf16 WMMA GEMM --------------------------
// C[M,N] f32 = A[M,K] bf16 @ B[K,N] bf16 (+ bias[N]).  M%64==0, N%64==0, K%32==0.
// Block = 256 thr = 8 waves, 64x64 tile; wave w: row_tile=w&3, col_half=w>>2.
__global__ __launch_bounds__(256) void k_gemm(const u16t* A, const u16t* Bm,
                                              float* C, const float* bias,
                                              int M, int N, int K) {
  __shared__ __align__(16) u16t As[64 * 32];      // row-major (stride 32)
  __shared__ __align__(16) u16t Bt[64 * 32];      // transposed: [col][k]
  int tid = threadIdx.x;
  int w = tid >> 5, lane = tid & 31;
  int rt = w & 3, ch = w >> 2;
  int m0 = blockIdx.y * 64, n0 = blockIdx.x * 64;
  int ia = tid * 8, ra = ia >> 5, ca = ia & 31;   // A: row ra, k ca..ca+7
  int ib = tid * 8, rb = ib >> 6, cb = ib & 63;   // B: k rb, col cb..cb+7
  v8f acc0 = {}, acc1 = {};
  for (int k0 = 0; k0 < K; k0 += 32) {
    // stage A tile (b128 per thread)
    *(u32x4*)(As + ia) = *(const u32x4*)(A + (size_t)(m0 + ra) * K + k0 + ca);
    // stage B tile transposed (b128 read, 8 scalar LDS stores)
    Pack8 pb;
    pb.q = *(const u32x4*)(Bm + (size_t)(k0 + rb) * N + n0 + cb);
#pragma unroll
    for (int i = 0; i < 8; ++i) Bt[(cb + i) * 32 + rb] = pb.s[i];
    // prefetch next K tile into cache while we compute
    if (k0 + 32 < K) {
      __builtin_prefetch(A + (size_t)(m0 + ra) * K + k0 + 32 + ca, 0, 1);
      __builtin_prefetch(Bm + (size_t)(k0 + 32 + rb) * N + n0 + cb, 0, 1);
    }
    __syncthreads();
    v16bf a  = frag_a_lds(As, rt * 16);
    v16bf b0 = frag_b_lds(Bt, ch * 32);
    v16bf b1 = frag_b_lds(Bt, ch * 32 + 16);
    acc0 = wmma_bf16(a, b0, acc0);
    acc1 = wmma_bf16(a, b1, acc1);
    __syncthreads();
  }
  FragF f0, f1; f0.v = acc0; f1.v = acc1;
  int mrow = m0 + rt * 16 + ((lane >> 4) << 3);
  int nc0 = n0 + ch * 32 + (lane & 15);
  int nc1 = nc0 + 16;
  float b0v = bias ? bias[nc0] : 0.f;
  float b1v = bias ? bias[nc1] : 0.f;
#pragma unroll
  for (int i = 0; i < 8; ++i) {
    C[(size_t)(mrow + i) * N + nc0] = f0.f[i] + b0v;
    C[(size_t)(mrow + i) * N + nc1] = f1.f[i] + b1v;
  }
}

// ------------------------- fused GRU step ----------------------------------
// rec = h @ Wh + br for all 3 gates on a 64-col slice, then GRU update.
// grid = U/64 blocks, 256 thr (8 waves); wave: rt=w&3 rows, ch=w>>2 col half.
// LDS: h tile (64x32) + 3 gate weight tiles transposed (192 cols x 32 k).
__global__ __launch_bounds__(256) void k_gru_step(
    const u16t* h_bf, const u16t* Wh, const float* br,
    const float* xp, int t, int L,
    const float* h_in, float* h_out, u16t* h_out_bf,
    float* seq_f32, u16t* seq_bf) {
  __shared__ __align__(16) u16t As[64 * 32];       // h tile, row-major
  __shared__ __align__(16) u16t Bt[192 * 32];      // [gate*64+col][k]
  int tid = threadIdx.x;
  int w = tid >> 5, lane = tid & 31;
  int rt = w & 3, ch = w >> 2;
  int ublk = blockIdx.x * 64;
  int ia = tid * 8, ra = ia >> 5, ca = ia & 31;
  int ib = tid * 8, rb = ib >> 6, cb = ib & 63;
  v8f z8 = {};
  v8f acc[3][2];
#pragma unroll
  for (int g = 0; g < 3; ++g)
#pragma unroll
    for (int c = 0; c < 2; ++c) acc[g][c] = z8;

  for (int k0 = 0; k0 < U_; k0 += 32) {
    *(u32x4*)(As + ia) = *(const u32x4*)(h_bf + (size_t)ra * U_ + k0 + ca);
#pragma unroll
    for (int g = 0; g < 3; ++g) {
      Pack8 pb;
      pb.q = *(const u32x4*)(Wh + (size_t)(k0 + rb) * (3 * U_) +
                             g * U_ + ublk + cb);
#pragma unroll
      for (int i = 0; i < 8; ++i) Bt[(g * 64 + cb + i) * 32 + rb] = pb.s[i];
    }
    __syncthreads();
    v16bf a = frag_a_lds(As, rt * 16);
#pragma unroll
    for (int g = 0; g < 3; ++g)
#pragma unroll
      for (int c = 0; c < 2; ++c) {
        v16bf b = frag_b_lds(Bt, g * 64 + ch * 32 + c * 16);
        acc[g][c] = wmma_bf16(a, b, acc[g][c]);
      }
    __syncthreads();
  }

  int mbase = rt * 16 + ((lane >> 4) << 3);
#pragma unroll
  for (int c = 0; c < 2; ++c) {
    FragF fz, fr, fh;
    fz.v = acc[0][c]; fr.v = acc[1][c]; fh.v = acc[2][c];
    int u = ublk + ch * 32 + c * 16 + (lane & 15);
    float brz = br[u], brr = br[U_ + u], brh = br[2 * U_ + u];
#pragma unroll
    for (int i = 0; i < 8; ++i) {
      int m = mbase + i;
      size_t xrow = ((size_t)m * L + t) * (size_t)(3 * U_);
      float xz = xp[xrow + u];
      float xr = xp[xrow + U_ + u];
      float xh = xp[xrow + 2 * U_ + u];
      float rz = fz.f[i] + brz, rr = fr.f[i] + brr, rh = fh.f[i] + brh;
      float zg = 1.f / (1.f + __expf(-(xz + rz)));
      float rg = 1.f / (1.f + __expf(-(xr + rr)));
      float hh = tanhf(xh + rg * rh);
      float ho = h_in[(size_t)m * U_ + u];
      float hn = zg * ho + (1.f - zg) * hh;
      h_out[(size_t)m * U_ + u] = hn;
      h_out_bf[(size_t)m * U_ + u] = f2bf(hn);
      if (seq_f32) {
        size_t so = ((size_t)m * L + t) * (size_t)U_ + u;
        seq_f32[so] = hn;
        seq_bf[so] = f2bf(hn);
      }
    }
  }
}

// ------------------------- attention / loss --------------------------------

__global__ __launch_bounds__(256) void k_att_scores(const float* q,
    const float* w2k, const float* att_v, const int* in_tok, float* scores) {
  __shared__ float red[256];
  int bs = blockIdx.x;        // b*S + s
  int b = bs >> 7;
  const float* qb = q + (size_t)b * U_;
  const float* wk = w2k + (size_t)bs * U_;
  float p = 0.f;
  for (int u = threadIdx.x; u < U_; u += 256)
    p += att_v[u] * tanhf(qb[u] + wk[u]);
  red[threadIdx.x] = p; __syncthreads();
  for (int s = 128; s > 0; s >>= 1) {
    if (threadIdx.x < s) red[threadIdx.x] += red[threadIdx.x + s];
    __syncthreads();
  }
  if (threadIdx.x == 0)
    scores[bs] = (in_tok[bs] != 0) ? red[0] : -1e9f;
}

__global__ __launch_bounds__(256) void k_softmax_ctx(const float* scores,
    const float* enc_out, const float* hn, u16t* ctx_hn_bf) {
  __shared__ float w[S_];
  int b = blockIdx.x;
  if (threadIdx.x < S_) w[threadIdx.x] = scores[b * S_ + threadIdx.x];
  __syncthreads();
  if (threadIdx.x == 0) {
    float mx = -1e30f;
    for (int s = 0; s < S_; ++s) mx = fmaxf(mx, w[s]);
    float sum = 0.f;
    for (int s = 0; s < S_; ++s) { w[s] = __expf(w[s] - mx); sum += w[s]; }
    float inv = 1.f / sum;
    for (int s = 0; s < S_; ++s) w[s] *= inv;
  }
  __syncthreads();
  for (int u = threadIdx.x; u < U_; u += 256) {
    float acc = 0.f;
    const float* eb = enc_out + ((size_t)b * S_) * U_ + u;
    for (int s = 0; s < S_; ++s) acc += w[s] * eb[(size_t)s * U_];
    ctx_hn_bf[(size_t)b * 2 * U_ + u]      = f2bf(acc);
    ctx_hn_bf[(size_t)b * 2 * U_ + U_ + u] = f2bf(hn[(size_t)b * U_ + u]);
  }
}

__global__ __launch_bounds__(256) void k_ce(const float* logits,
                                            const int* tgt, int t, float* loss) {
  __shared__ float red[256];
  int b = blockIdx.x;
  const float* lr = logits + (size_t)b * V_;
  float mx = -1e30f;
  for (int v = threadIdx.x; v < V_; v += 256) mx = fmaxf(mx, lr[v]);
  red[threadIdx.x] = mx; __syncthreads();
  for (int s = 128; s > 0; s >>= 1) {
    if (threadIdx.x < s)
      red[threadIdx.x] = fmaxf(red[threadIdx.x], red[threadIdx.x + s]);
    __syncthreads();
  }
  mx = red[0]; __syncthreads();
  float p = 0.f;
  for (int v = threadIdx.x; v < V_; v += 256) p += __expf(lr[v] - mx);
  red[threadIdx.x] = p; __syncthreads();
  for (int s = 128; s > 0; s >>= 1) {
    if (threadIdx.x < s) red[threadIdx.x] += red[threadIdx.x + s];
    __syncthreads();
  }
  if (threadIdx.x == 0) {
    int y = tgt[b * T_ + t + 1];
    if (y != 0) {
      float lse = __logf(red[0]) + mx;
      atomicAdd(loss, lse - lr[y]);
    }
  }
}

__global__ __launch_bounds__(256) void k_finalize(const float* loss,
                                                  const int* tgt, float* out) {
  __shared__ int red[256];
  int c = 0;
  for (int i = threadIdx.x; i < B_ * T_; i += 256) c += (tgt[i] != 0);
  red[threadIdx.x] = c; __syncthreads();
  for (int s = 128; s > 0; s >>= 1) {
    if (threadIdx.x < s) red[threadIdx.x] += red[threadIdx.x + s];
    __syncthreads();
  }
  if (threadIdx.x == 0) out[0] = loss[0] / (float)red[0];
}

// ------------------------------- host --------------------------------------

extern "C" void kernel_launch(void* const* d_in, const int* in_sizes, int n_in,
                              void* d_out, int out_size, void* d_ws,
                              size_t ws_size, hipStream_t stream) {
  (void)in_sizes; (void)n_in; (void)out_size; (void)ws_size;
  const int*   in_tok  = (const int*)  d_in[0];
  const int*   tgt_tok = (const int*)  d_in[1];
  const float* enc_emb = (const float*)d_in[2];
  const float* enc_Wx  = (const float*)d_in[3];
  const float* enc_Wh  = (const float*)d_in[4];
  const float* enc_bi  = (const float*)d_in[5];
  const float* enc_br  = (const float*)d_in[6];
  const float* dec_emb = (const float*)d_in[7];
  const float* dec_Wx  = (const float*)d_in[8];
  const float* dec_Wh  = (const float*)d_in[9];
  const float* dec_bi  = (const float*)d_in[10];
  const float* dec_br  = (const float*)d_in[11];
  const float* att_W1  = (const float*)d_in[12];
  const float* att_W2  = (const float*)d_in[13];
  const float* att_v   = (const float*)d_in[14];
  const float* conv_W  = (const float*)d_in[15];
  const float* out_W   = (const float*)d_in[16];
  const float* out_b   = (const float*)d_in[17];

  char* wsb = (char*)d_ws; size_t off = 0;
  auto alloc = [&](size_t bytes) -> void* {
    void* p = wsb + off;
    off += (bytes + 255) & ~(size_t)255;
    return p;
  };
  const int L = T_ - 1;  // 127 decoder steps
  u16t* encWx_bf = (u16t*)alloc((size_t)E_ * 3 * U_ * 2);
  u16t* encWh_bf = (u16t*)alloc((size_t)U_ * 3 * U_ * 2);
  u16t* decWx_bf = (u16t*)alloc((size_t)E_ * 3 * U_ * 2);
  u16t* decWh_bf = (u16t*)alloc((size_t)U_ * 3 * U_ * 2);
  u16t* attW1_bf = (u16t*)alloc((size_t)U_ * U_ * 2);
  u16t* attW2_bf = (u16t*)alloc((size_t)U_ * U_ * 2);
  u16t* convW_bf = (u16t*)alloc((size_t)2 * U_ * U_ * 2);
  u16t* outW_bf  = (u16t*)alloc((size_t)U_ * V_ * 2);
  u16t* Xenc     = (u16t*)alloc((size_t)B_ * S_ * E_ * 2);
  u16t* Xdec     = (u16t*)alloc((size_t)B_ * L  * E_ * 2);
  float* xp_enc  = (float*)alloc((size_t)B_ * S_ * 3 * U_ * 4);
  float* xp_dec  = (float*)alloc((size_t)B_ * L  * 3 * U_ * 4);
  float* enc_out = (float*)alloc((size_t)B_ * S_ * U_ * 4);
  u16t* enc_out_bf = (u16t*)alloc((size_t)B_ * S_ * U_ * 2);
  float* w2k     = (float*)alloc((size_t)B_ * S_ * U_ * 4);
  float* hf0     = (float*)alloc((size_t)B_ * U_ * 4);
  float* hf1     = (float*)alloc((size_t)B_ * U_ * 4);
  u16t* hb0      = (u16t*)alloc((size_t)B_ * U_ * 2);
  u16t* hb1      = (u16t*)alloc((size_t)B_ * U_ * 2);
  float* q       = (float*)alloc((size_t)B_ * U_ * 4);
  float* scores  = (float*)alloc((size_t)B_ * S_ * 4);
  u16t* ctxhn    = (u16t*)alloc((size_t)B_ * 2 * U_ * 2);
  float* av      = (float*)alloc((size_t)B_ * U_ * 4);
  u16t* av_bf    = (u16t*)alloc((size_t)B_ * U_ * 2);
  float* logits  = (float*)alloc((size_t)B_ * V_ * 4);
  float* loss    = (float*)alloc(4);
  float* hf[2] = { hf0, hf1 };
  u16t*  hb[2] = { hb0, hb1 };

  auto cvt = [&](const float* s, u16t* d, size_t n) {
    k_cvt<<<dim3((unsigned)((n + 255) / 256)), dim3(256), 0, stream>>>(s, d, (int)n);
  };
  auto gemm = [&](const u16t* A, const u16t* Bm, float* C, const float* bias,
                  int M, int N, int K) {
    k_gemm<<<dim3((unsigned)(N / 64), (unsigned)(M / 64)), dim3(256), 0, stream>>>(
        A, Bm, C, bias, M, N, K);
  };

  // ---- one-time conversions / gathers ----
  cvt(enc_Wx, encWx_bf, (size_t)E_ * 3 * U_);
  cvt(enc_Wh, encWh_bf, (size_t)U_ * 3 * U_);
  cvt(dec_Wx, decWx_bf, (size_t)E_ * 3 * U_);
  cvt(dec_Wh, decWh_bf, (size_t)U_ * 3 * U_);
  cvt(att_W1, attW1_bf, (size_t)U_ * U_);
  cvt(att_W2, attW2_bf, (size_t)U_ * U_);
  cvt(conv_W, convW_bf, (size_t)2 * U_ * U_);
  cvt(out_W,  outW_bf,  (size_t)U_ * V_);

  k_gather<<<dim3((B_ * S_ * E_ + 255) / 256), dim3(256), 0, stream>>>(
      in_tok, enc_emb, Xenc, B_ * S_, S_, S_);
  k_gather<<<dim3((B_ * L * E_ + 255) / 256), dim3(256), 0, stream>>>(
      tgt_tok, dec_emb, Xdec, B_ * L, L, T_);

  // xp = X @ Wx + bi (batched over all timesteps)
  gemm(Xenc, encWx_bf, xp_enc, enc_bi, B_ * S_, 3 * U_, E_);
  gemm(Xdec, decWx_bf, xp_dec, dec_bi, B_ * L,  3 * U_, E_);

  // ---- init state / loss ----
  k_zero_f32<<<dim3((B_ * U_ + 255) / 256), dim3(256), 0, stream>>>(hf[0], B_ * U_);
  k_zero_u16<<<dim3((B_ * U_ + 255) / 256), dim3(256), 0, stream>>>(hb[0], B_ * U_);
  k_zero_f32<<<dim3(1), dim3(256), 0, stream>>>(loss, 1);

  // ---- encoder scan (128 steps) ----
  for (int t = 0; t < S_; ++t) {
    k_gru_step<<<dim3(U_ / 64), dim3(256), 0, stream>>>(
        hb[t & 1], encWh_bf, enc_br, xp_enc, t, S_,
        hf[t & 1], hf[(t + 1) & 1], hb[(t + 1) & 1],
        enc_out, enc_out_bf);
  }
  // after 128 steps the final state is in slot 0 (128 & 1 == 0)

  // w2k = enc_out @ att_W2
  gemm(enc_out_bf, attW2_bf, w2k, nullptr, B_ * S_, U_, U_);

  // ---- decoder scan (127 steps) ----
  for (int t = 0; t < L; ++t) {
    k_gru_step<<<dim3(U_ / 64), dim3(256), 0, stream>>>(
        hb[t & 1], decWh_bf, dec_br, xp_dec, t, L,
        hf[t & 1], hf[(t + 1) & 1], hb[(t + 1) & 1],
        nullptr, nullptr);
    float* hn_f = hf[(t + 1) & 1];
    u16t*  hn_b = hb[(t + 1) & 1];

    gemm(hn_b, attW1_bf, q, nullptr, B_, U_, U_);                     // q = hn @ W1
    k_att_scores<<<dim3(B_ * S_), dim3(256), 0, stream>>>(q, w2k, att_v,
                                                          in_tok, scores);
    k_softmax_ctx<<<dim3(B_), dim3(256), 0, stream>>>(scores, enc_out,
                                                      hn_f, ctxhn);
    gemm(ctxhn, convW_bf, av, nullptr, B_, U_, 2 * U_);               // concat @ conv_W
    k_tanh_bf<<<dim3((B_ * U_ + 255) / 256), dim3(256), 0, stream>>>(av, av_bf,
                                                                     B_ * U_);
    gemm(av_bf, outW_bf, logits, out_b, B_, V_, U_);                  // logits
    k_ce<<<dim3(B_), dim3(256), 0, stream>>>(logits, tgt_tok, t, loss);
  }

  k_finalize<<<dim3(1), dim3(256), 0, stream>>>(loss, tgt_tok, (float*)d_out);
}
